// FusedAttention_13443247637069
// MI455X (gfx1250) — matrix operand order repaired and checked by reading, not compiled, not run on hardware
//
#include <hip/hip_runtime.h>
#include <hip/hip_bf16.h>

// ---------------------------------------------------------------------------
// CDNA5 (gfx1250) wave32 WMMA types
// ---------------------------------------------------------------------------
typedef __attribute__((ext_vector_type(16))) __bf16 v16bf;
typedef __attribute__((ext_vector_type(8)))  float  v8f;
typedef __attribute__((ext_vector_type(4)))  float  vf4;

#define WMMA_BF16(A, B, C) \
  __builtin_amdgcn_wmma_f32_16x16x32_bf16(false, (A), false, (B), (short)0, (C), false, false)

// gfx1250 async global->LDS DMA (probe-guarded; fallback = synchronous path)
#if __has_builtin(__builtin_amdgcn_global_load_async_to_lds_b128)
#define USE_ASYNC 1
#else
#define USE_ASYNC 0
#endif

#if __has_builtin(__builtin_amdgcn_s_wait_asynccnt)
#define WAIT_ASYNC() __builtin_amdgcn_s_wait_asynccnt(0)
#else
#define WAIT_ASYNC() asm volatile("s_wait_asynccnt 0" ::: "memory")
#endif

// Exact pointee type the builtin wants: int __attribute__((vector_size(16)))
typedef int v4i_ls __attribute__((vector_size(16)));
typedef __attribute__((address_space(1))) v4i_ls* gptr128_t;
typedef __attribute__((address_space(3))) v4i_ls* lptr128_t;

// Model constants
#define BATCH 32
#define PAST  2048
#define H_Q   32
#define H_KV  8
#define DHEAD 128
#define HID   4096
#define SLOTS (BATCH * (PAST + 1))          // 65568
#define KV_ROW (2 * H_KV * DHEAD)           // 2048 floats per slot

__device__ __forceinline__ void cvt4(v16bf& d, int base, vf4 v) {
  d[base + 0] = (__bf16)v[0];
  d[base + 1] = (__bf16)v[1];
  d[base + 2] = (__bf16)v[2];
  d[base + 3] = (__bf16)v[3];
}

__device__ __forceinline__ unsigned pack_bf2(float a, float b) {
  __bf16 ha = (__bf16)a, hb = (__bf16)b;
  unsigned short ua = __builtin_bit_cast(unsigned short, ha);
  unsigned short ub = __builtin_bit_cast(unsigned short, hb);
  return (unsigned)ua | ((unsigned)ub << 16);
}

// ---------------------------------------------------------------------------
// Kernel 1: Y[32,N] = X[32,K] * W[N,K]^T   (bf16 WMMA, f32 accum)
// Block = 256 threads = 8 waves; wave w owns 16 output columns.
// A fragment (16x32 bf16):  lane m=L&15, h=L>>4, elem e -> K = 8h+e (e<8) else 16+8h+(e-8)
// B fragment (32x16 bf16):  lane n=L&15, h=L>>4, elem e -> K = 8h+e (e<8) else 16+8h+(e-8)
// C/D (16x16 f32):          lane n=L&15, vgpr r -> row m = 8*(L>>4)+r
// ---------------------------------------------------------------------------
__global__ __launch_bounds__(256) void gemm32_bf16_wmma(
    const float* __restrict__ X, const float* __restrict__ W,
    float* __restrict__ Y, int N, int K)
{
  const int wave = threadIdx.x >> 5;
  const int lane = threadIdx.x & 31;
  const int nidx = lane & 15;
  const int hh   = lane >> 4;
  const int n    = blockIdx.x * 128 + wave * 16 + nidx;

  const float* wrow = W + n * K;          // <= 25M elements: int math is safe
  const float* xr0  = X + nidx * K;
  const float* xr1  = X + (nidx + 16) * K;

  v8f c0 = {}, c1 = {};
  for (int k0 = 0; k0 < K; k0 += 32) {
    __builtin_prefetch(wrow + k0 + 1024, 0, 1);   // -> global_prefetch_b8

    v16bf bm;
    {
      const vf4* s = (const vf4*)(wrow + k0 + 8 * hh);
      const vf4* t = (const vf4*)(wrow + k0 + 16 + 8 * hh);
      cvt4(bm, 0, s[0]); cvt4(bm, 4, s[1]); cvt4(bm, 8, t[0]); cvt4(bm, 12, t[1]);
    }
    v16bf a0, a1;
    {
      const vf4* lo = (const vf4*)(xr0 + k0 + 8 * hh);
      const vf4* hi = (const vf4*)(xr0 + k0 + 16 + 8 * hh);
      cvt4(a0, 0, lo[0]); cvt4(a0, 4, lo[1]); cvt4(a0, 8, hi[0]); cvt4(a0, 12, hi[1]);
    }
    {
      const vf4* lo = (const vf4*)(xr1 + k0 + 8 * hh);
      const vf4* hi = (const vf4*)(xr1 + k0 + 16 + 8 * hh);
      cvt4(a1, 0, lo[0]); cvt4(a1, 4, lo[1]); cvt4(a1, 8, hi[0]); cvt4(a1, 12, hi[1]);
    }
    c0 = WMMA_BF16(a0, bm, c0);
    c1 = WMMA_BF16(a1, bm, c1);
  }
  #pragma unroll
  for (int r = 0; r < 8; ++r) {
    int m = 8 * hh + r;
    Y[m * N + n]        = c0[r];
    Y[(m + 16) * N + n] = c1[r];
  }
}

// ---------------------------------------------------------------------------
// Kernel 2: RoPE on xq (32x32x128) and xk (32x8x128), in place
// ---------------------------------------------------------------------------
__global__ void rope_kernel(float* __restrict__ xq, float* __restrict__ xk,
                            const float* __restrict__ cosb,
                            const float* __restrict__ sinb)
{
  int tid = blockIdx.x * blockDim.x + threadIdx.x;   // 1280 rows * 64 pairs
  int row = tid >> 6;
  int d   = tid & 63;
  float* buf; int b;
  if (row < BATCH * H_Q) { buf = xq + row * DHEAD; b = row >> 5; }
  else { int rk = row - BATCH * H_Q; buf = xk + rk * DHEAD; b = rk >> 3; }
  float v1 = buf[d], v2 = buf[d + 64];
  float c1 = cosb[b * DHEAD + d], c2 = cosb[b * DHEAD + d + 64];
  float s1 = sinb[b * DHEAD + d], s2 = sinb[b * DHEAD + d + 64];
  buf[d]      = v1 * c1 - v2 * s1;
  buf[d + 64] = v2 * c2 + v1 * s2;
}

// ---------------------------------------------------------------------------
// Kernel 3: streaming copy of kv_buffer -> out_kv (non-temporal f32x4)
// ---------------------------------------------------------------------------
__global__ void copy_kv_kernel(const vf4* __restrict__ src, vf4* __restrict__ dst, long n4)
{
  long i = (long)blockIdx.x * blockDim.x + threadIdx.x;
  long stride = (long)gridDim.x * blockDim.x;
  for (; i < n4; i += stride) {
    vf4 v = __builtin_nontemporal_load(&src[i]);
    __builtin_nontemporal_store(v, &dst[i]);
  }
}

// ---------------------------------------------------------------------------
// Kernel 4: scatter new (rotated K, V) into the 32 decode slots of out_kv
// ---------------------------------------------------------------------------
__global__ void scatter_kernel(float* __restrict__ outkv,
                               const float* __restrict__ xk,
                               const float* __restrict__ xv,
                               const int* __restrict__ dec)
{
  int b = blockIdx.x;
  long slot = dec[b];
  for (int i = threadIdx.x; i < KV_ROW; i += blockDim.x) {
    int h = i >> 7, d = i & 127;
    float v = (h < H_KV) ? xk[(b * H_KV + h) * DHEAD + d]
                         : xv[(b * H_KV + (h - H_KV)) * DHEAD + d];
    outkv[slot * KV_ROW + i] = v;
  }
}

// ---------------------------------------------------------------------------
// Kernel 5: flash-decode attention.
// Grid = 256 blocks, one per (b, kv-head). Block = 256 threads = 8 waves.
// Each wave owns key chunks of 32 (chunk = wave, wave+8, ... over 65 chunks
// covering 2049 keys), runs online softmax with WMMA for both QK^T and PV.
// V rows are staged to LDS with GLOBAL_LOAD_ASYNC_TO_LDS_B128 when the
// toolchain exposes it (overlaps the gather DMA with the QK^T WMMAs; the
// wave only waits on ASYNCcnt right before the PV fragment reads).
// ---------------------------------------------------------------------------
#define AW 8   // waves per block

#if USE_ASYNC
typedef float svt;          // async DMA copies raw f32; convert at frag build
#else
typedef __bf16 svt;         // synchronous path stages pre-converted bf16
#endif

__global__ __launch_bounds__(256) void attn_kernel(
    const float* __restrict__ xq,   // 32x32x128 (rope'd)
    const float* __restrict__ xk,   // 32x8x128  (rope'd)
    const float* __restrict__ xv,   // 32x8x128
    const float* __restrict__ kv,   // kv_buffer (SLOTS x 16 x 128)
    const int*  __restrict__ sel,   // 32x2048
    float* __restrict__ o)          // 32x32x128
{
  __shared__ __align__(16) __bf16 sP[AW][16][32];    // probs transpose buffer
  __shared__ __align__(16) svt    sV[AW][32][DHEAD]; // staged V chunk
  __shared__ float sO[AW][4][DHEAD];
  __shared__ float sM[AW][4];
  __shared__ float sL[AW][4];

  const int bg   = blockIdx.x;
  const int b    = bg >> 3, g = bg & 7;
  const int wave = threadIdx.x >> 5;
  const int lane = threadIdx.x & 31;
  const int nidx = lane & 15;
  const int hh   = lane >> 4;
  const float scale = 0.08838834764831845f;  // 1/sqrt(128)

  // ---- Q fragments (rows 0..3 real, rows 4..15 zero), scale folded in ----
  v16bf qa[4];
  {
    int m = nidx;
    const float* qrow = xq + (b * H_Q + g * 4 + (m < 4 ? m : 0)) * DHEAD;
    #pragma unroll
    for (int kb = 0; kb < 4; ++kb) {
      const vf4* lo = (const vf4*)(qrow + kb * 32 + 8 * hh);
      const vf4* hi = (const vf4*)(qrow + kb * 32 + 16 + 8 * hh);
      vf4 l0 = lo[0], l1 = lo[1], h0 = hi[0], h1 = hi[1];
      v16bf a;
      #pragma unroll
      for (int e = 0; e < 4; ++e) {
        a[e]      = (__bf16)((m < 4 ? l0[e] : 0.f) * scale);
        a[4 + e]  = (__bf16)((m < 4 ? l1[e] : 0.f) * scale);
        a[8 + e]  = (__bf16)((m < 4 ? h0[e] : 0.f) * scale);
        a[12 + e] = (__bf16)((m < 4 ? h1[e] : 0.f) * scale);
      }
      qa[kb] = a;
    }
  }

  v8f oacc[8];
  #pragma unroll
  for (int j = 0; j < 8; ++j) { v8f z = {}; oacc[j] = z; }
  float rowM[4] = {-1e30f, -1e30f, -1e30f, -1e30f};
  float rowL[4] = {0.f, 0.f, 0.f, 0.f};

  for (int chunk = wave; chunk <= 64; chunk += AW) {
    const int kbase = chunk * 32;

    // ---- stage V chunk to LDS (one gathered key row per lane) ----
    {
      int key = kbase + lane;
      int ck  = key <= PAST ? key : PAST;
      const float* vrow = (ck < PAST)
          ? (kv + (long)sel[b * PAST + ck] * KV_ROW + (H_KV + g) * DHEAD)
          : (xv + (b * H_KV + g) * DHEAD);
#if USE_ASYNC
      #pragma unroll
      for (int i = 0; i < DHEAD; i += 4) {
        __builtin_amdgcn_global_load_async_to_lds_b128(
            (gptr128_t)(vrow + i),
            (lptr128_t)(&sV[wave][lane][i]),
            0, 0);
      }
#else
      #pragma unroll
      for (int i = 0; i < DHEAD; i += 4) {
        vf4 vv = *(const vf4*)(vrow + i);
        uint2 pk;
        pk.x = pack_bf2(vv[0], vv[1]);
        pk.y = pack_bf2(vv[2], vv[3]);
        *(uint2*)(&sV[wave][lane][i]) = pk;
      }
#endif
    }

    // ---- scores: two 16-key tiles via WMMA over D=128 (overlaps async DMA) ----
    v8f s0 = {}, s1 = {};
    #pragma unroll
    for (int tt = 0; tt < 2; ++tt) {
      int key = kbase + tt * 16 + nidx;
      int ck  = key <= PAST ? key : PAST;
      const float* krow = (ck < PAST)
          ? (kv + (long)sel[b * PAST + ck] * KV_ROW + g * DHEAD)
          : (xk + (b * H_KV + g) * DHEAD);
      v8f c = {};
      #pragma unroll
      for (int kb = 0; kb < 4; ++kb) {
        const vf4* s4 = (const vf4*)(krow + kb * 32 + 8 * hh);
        const vf4* t4 = (const vf4*)(krow + kb * 32 + 16 + 8 * hh);
        v16bf bm;
        cvt4(bm, 0, s4[0]); cvt4(bm, 4, s4[1]); cvt4(bm, 8, t4[0]); cvt4(bm, 12, t4[1]);
        c = WMMA_BF16(qa[kb], bm, c);
      }
      if (key > PAST) {
        #pragma unroll
        for (int r = 0; r < 8; ++r) c[r] = -1e30f;  // mask out-of-range keys
      }
      if (tt == 0) s0 = c; else s1 = c;
    }

    // ---- online softmax stats (rows live in vgprs 0..3 of lanes 0..15) ----
    float corr[4], nm[4];
    #pragma unroll
    for (int r = 0; r < 4; ++r) {
      float t = fmaxf(s0[r], s1[r]);
      #pragma unroll
      for (int off = 8; off; off >>= 1) t = fmaxf(t, __shfl_xor(t, off, 32));
      float newm = fmaxf(rowM[r], t);
      corr[r] = __expf(rowM[r] - newm);
      rowM[r] = newm;
      nm[r]   = newm;
    }
    float p0[8], p1[8];
    #pragma unroll
    for (int r = 0; r < 8; ++r) {
      bool valid = (hh == 0) && (r < 4);
      p0[r] = valid ? __expf(s0[r] - nm[r & 3]) : 0.f;
      p1[r] = valid ? __expf(s1[r] - nm[r & 3]) : 0.f;
    }
    #pragma unroll
    for (int r = 0; r < 4; ++r) {
      float t = p0[r] + p1[r];
      #pragma unroll
      for (int off = 8; off; off >>= 1) t += __shfl_xor(t, off, 32);
      rowL[r] = rowL[r] * corr[r] + t;
    }
    // rescale previous accumulators (rows 4..7 are zero; scale is harmless)
    #pragma unroll
    for (int j = 0; j < 8; ++j) {
      v8f t = oacc[j];
      #pragma unroll
      for (int r = 0; r < 8; ++r) t[r] *= corr[r & 3];
      oacc[j] = t;
    }

    // ---- transpose probs through LDS into an A fragment ----
    #pragma unroll
    for (int r = 0; r < 8; ++r) {
      int m = 8 * hh + r;
      sP[wave][m][nidx]      = (__bf16)p0[r];
      sP[wave][m][16 + nidx] = (__bf16)p1[r];
    }
    v16bf pa;
    {
      int m = nidx;
      #pragma unroll
      for (int e = 0; e < 16; ++e) {
        int k = (e < 8) ? (8 * hh + e) : (16 + 8 * hh + (e - 8));
        pa[e] = sP[wave][m][k];
      }
    }

#if USE_ASYNC
    WAIT_ASYNC();   // V DMA must have landed before the fragment reads
#endif

    // ---- P x V : 8 WMMAs (one per 16-wide d tile), V from LDS ----
    #pragma unroll
    for (int j = 0; j < 8; ++j) {
      int d = j * 16 + nidx;
      v16bf vb;
      #pragma unroll
      for (int e = 0; e < 16; ++e) vb[e] = (__bf16)sV[wave][(e < 8) ? (8 * hh + e) : (16 + 8 * hh + (e - 8))][d];
      oacc[j] = WMMA_BF16(pa, vb, oacc[j]);
    }
  }

  // ---- per-wave partials -> LDS ----
  if (hh == 0) {
    if (nidx == 0) {
      #pragma unroll
      for (int r = 0; r < 4; ++r) { sM[wave][r] = rowM[r]; sL[wave][r] = rowL[r]; }
    }
    #pragma unroll
    for (int j = 0; j < 8; ++j)
      #pragma unroll
      for (int r = 0; r < 4; ++r)
        sO[wave][r][j * 16 + nidx] = oacc[j][r];
  }
  __syncthreads();

  // ---- combine waves, normalize, store o ----
  if (threadIdx.x < DHEAD) {
    int d = threadIdx.x;
    #pragma unroll
    for (int r = 0; r < 4; ++r) {
      float M = sM[0][r];
      #pragma unroll
      for (int w = 1; w < AW; ++w) M = fmaxf(M, sM[w][r]);
      float L = 0.f, acc = 0.f;
      #pragma unroll
      for (int w = 0; w < AW; ++w) {
        float f = __expf(sM[w][r] - M);
        L   += f * sL[w][r];
        acc += f * sO[w][r][d];
      }
      o[(b * H_Q + g * 4 + r) * DHEAD + d] = acc / L;
    }
  }
}

// ---------------------------------------------------------------------------
// Launch
// ---------------------------------------------------------------------------
extern "C" void kernel_launch(void* const* d_in, const int* in_sizes, int n_in,
                              void* d_out, int out_size, void* d_ws, size_t ws_size,
                              hipStream_t stream) {
  (void)in_sizes; (void)n_in; (void)out_size; (void)ws_size;

  const float* x    = (const float*)d_in[0];
  const float* wq   = (const float*)d_in[1];
  const float* wk   = (const float*)d_in[2];
  const float* wv   = (const float*)d_in[3];
  const float* wo   = (const float*)d_in[4];
  const float* cosb = (const float*)d_in[5];
  const float* sinb = (const float*)d_in[6];
  const float* kvb  = (const float*)d_in[7];
  const int*   sel  = (const int*)d_in[8];
  const int*   dec  = (const int*)d_in[9];

  float* out   = (float*)d_out;                 // 32*4096
  float* outkv = out + BATCH * HID;             // SLOTS*16*128

  float* ws  = (float*)d_ws;
  float* xqw = ws;                              // 131072
  float* xkw = xqw + BATCH * H_Q * DHEAD;       // 32768
  float* xvw = xkw + BATCH * H_KV * DHEAD;      // 32768
  float* ow  = xvw + BATCH * H_KV * DHEAD;      // 131072

  // QKV projections (WMMA bf16)
  gemm32_bf16_wmma<<<32, 256, 0, stream>>>(x, wq, xqw, H_Q * DHEAD, HID);
  gemm32_bf16_wmma<<<8,  256, 0, stream>>>(x, wk, xkw, H_KV * DHEAD, HID);
  gemm32_bf16_wmma<<<8,  256, 0, stream>>>(x, wv, xvw, H_KV * DHEAD, HID);

  // RoPE
  rope_kernel<<<(BATCH * (H_Q + H_KV) * 64) / 256, 256, 0, stream>>>(xqw, xkw, cosb, sinb);

  // kv_new = copy of kv_buffer + scatter of the 32 decode slots
  long n4 = (long)SLOTS * KV_ROW / 4;
  copy_kv_kernel<<<4096, 256, 0, stream>>>((const vf4*)kvb, (vf4*)outkv, n4);
  scatter_kernel<<<BATCH, 256, 0, stream>>>(outkv, xkw, xvw, dec);

  // attention (flash decode, WMMA + async V staging)
  attn_kernel<<<BATCH * H_KV, 256, 0, stream>>>(xqw, xkw, xvw, kvb, sel, ow);

  // output projection (WMMA bf16)
  gemm32_bf16_wmma<<<32, 256, 0, stream>>>(ow, wo, out, HID, H_Q * DHEAD);
}
